// LocalMemoryDecoder_85306640433372
// MI455X (gfx1250) — matrix operand order repaired and checked by reading, not compiled, not run on hardware
//
#include <hip/hip_runtime.h>
#include <hip/hip_bf16.h>

typedef __bf16 bf16_t;
typedef __attribute__((ext_vector_type(16))) __bf16 v16bf;
typedef __attribute__((ext_vector_type(8)))  __bf16 v8bf;
typedef __attribute__((ext_vector_type(8)))  float  v8f;
typedef __attribute__((__vector_size__(4 * sizeof(int)))) int v4i_t;
typedef __attribute__((address_space(1))) v4i_t gv4i_t;   // global int4
typedef __attribute__((address_space(3))) v4i_t lv4i_t;   // LDS int4

// Problem constants (match reference)
constexpr int B = 32, T = 64, V = 16000, E = 256, H = 256;
constexpr int LH = 128, LKB = 256, LKE = 256, S = 256, SOS = 1;
constexpr int H3 = 3 * H;  // 768
constexpr int WPB = 4;     // waves per block in the M=32 LDS-staged GEMM

// ---------------------------------------------------------------------------
// WMMA fragment loaders (bf16, 16x16x32).
// A (16x32, MxK): lane row = lane%16; lo lanes hold K {0..15},
//                 hi lanes hold K {16..31}  -> two 16B loads.
// B (32x16, KxN) from row-major W[N,K]: lane row n = lane%16; lo lanes hold
//                 K {0..15}, hi lanes K {16..31} -> two contiguous 16B loads.
// ---------------------------------------------------------------------------
__device__ __forceinline__ v16bf cat8(v8bf lo, v8bf hi) {
  return __builtin_shufflevector(lo, hi, 0, 1, 2, 3, 4, 5, 6, 7, 8, 9, 10, 11,
                                 12, 13, 14, 15);
}

__device__ __forceinline__ v16bf load_fragA(const bf16_t* row, int k0, int lane) {
  const bf16_t* p = row + k0 + ((lane & 16) ? 16 : 0);
  return cat8(*(const v8bf*)p, *(const v8bf*)(p + 8));
}

__device__ __forceinline__ v16bf load_fragB(const bf16_t* row, int k0, int lane) {
  const bf16_t* p = row + k0 + ((lane & 16) ? 16 : 0);
  return cat8(*(const v8bf*)p, *(const v8bf*)(p + 8));
}

__device__ __forceinline__ float actf(float v, int act) {
  if (act == 1) return tanhf(v);
  if (act == 2) return fmaxf(v, 0.f);
  return v;
}

// ---------------------------------------------------------------------------
// Generic batched GEMM (large M):  Out[b] = act(A[b](MxK) * W[b](NxK)^T + bias)
// 2 m-tiles x 4 n-tiles per wave (M%32==0, N%64==0, K%32==0).
// act: 0=none 1=tanh 2=relu.  Optional strided bf16 output copy.
// ---------------------------------------------------------------------------
__global__ void wmma_gemm_kernel(
    const bf16_t* __restrict__ A, long long sA, int lda,
    const bf16_t* __restrict__ W, long long sW, int ldw,
    const float* __restrict__ bias, long long sBias,
    float* __restrict__ Out, long long sO, int ldo,
    bf16_t* __restrict__ OutBf, long long sOB, int ldob,
    int M, int N, int K, int act, int wavesTotal) {
  int wave = (int)((blockIdx.x * blockDim.x + threadIdx.x) >> 5);
  if (wave >= wavesTotal) return;
  int lane = threadIdx.x & 31;
  int ng = N >> 6;              // groups of 4 n-tiles
  int perB = (M >> 5) * ng;     // groups of 2 m-tiles
  int batch = wave / perB;
  int tt = wave - batch * perB;
  int mg = tt / ng;
  int ngi = tt - mg * ng;

  const bf16_t* arow0 =
      A + batch * sA + (long long)(mg * 32 + (lane & 15)) * lda;
  const bf16_t* arow1 = arow0 + (long long)16 * lda;
  const bf16_t* wrow0 =
      W + batch * sW + (long long)(ngi * 64 + (lane & 15)) * ldw;

  v8f z = {0.f, 0.f, 0.f, 0.f, 0.f, 0.f, 0.f, 0.f};
  v8f acc[8];
#pragma unroll
  for (int i = 0; i < 8; ++i) acc[i] = z;

  for (int k0 = 0; k0 < K; k0 += 32) {
    v16bf a0 = load_fragA(arow0, k0, lane);
    v16bf a1 = load_fragA(arow1, k0, lane);
#pragma unroll
    for (int j = 0; j < 4; ++j) {
      v16bf bv = load_fragB(wrow0 + (long long)j * 16 * ldw, k0, lane);
      acc[j] = __builtin_amdgcn_wmma_f32_16x16x32_bf16(false, a0, false, bv,
                                                       (short)0, acc[j], false,
                                                       false);
      acc[4 + j] = __builtin_amdgcn_wmma_f32_16x16x32_bf16(
          false, a1, false, bv, (short)0, acc[4 + j], false, false);
    }
  }

  float* outp = Out + batch * sO;
  bf16_t* outbf = OutBf ? (OutBf + batch * sOB) : (bf16_t*)0;
#pragma unroll
  for (int mi = 0; mi < 2; ++mi) {
    int mbase = mg * 32 + mi * 16 + ((lane & 16) ? 8 : 0);
#pragma unroll
    for (int j = 0; j < 4; ++j) {
      int n = ngi * 64 + j * 16 + (lane & 15);
      float bb = bias ? bias[batch * sBias + n] : 0.f;
#pragma unroll
      for (int r = 0; r < 8; ++r) {
        float v = actf(acc[mi * 4 + j][r] + bb, act);
        long long m = mbase + r;
        outp[m * ldo + n] = v;
        if (outbf) outbf[m * ldob + n] = (bf16_t)v;
      }
    }
  }
}

// ---------------------------------------------------------------------------
// M=32 GEMM with A staged in LDS via CDNA5 async global->LDS copy.
// Block = 4 waves; the whole A (32xK bf16, K<=768 -> <=48KB) is loaded once
// per workgroup with GLOBAL_LOAD_ASYNC_TO_LDS_B128 (ASYNCcnt), then each wave
// computes 2 m-tiles x 4 n-tiles reading A fragments from LDS (ds_load_b128)
// while streaming B from global/L2.
// ---------------------------------------------------------------------------
__global__ void wmma_gemm_m32_kernel(
    const bf16_t* __restrict__ A, long long sA, int lda,
    const bf16_t* __restrict__ W, long long sW, int ldw,
    const float* __restrict__ bias, long long sBias,
    float* __restrict__ Out, long long sO, int ldo,
    bf16_t* __restrict__ OutBf, long long sOB, int ldob,
    int N, int K, int act, int blocksPerBatch) {
  extern __shared__ bf16_t As[];  // compact [32, K]
  int tid = threadIdx.x;          // WPB*32 threads
  int wave = tid >> 5;
  int lane = tid & 31;
  int batch = blockIdx.x / blocksPerBatch;
  int bg = blockIdx.x - batch * blocksPerBatch;
  const bf16_t* Ab = A + batch * sA;

  int vpr = K >> 3;        // 16B vectors per row
  int nvec = 32 * vpr;     // total 16B vectors in A
#if __has_builtin(__builtin_amdgcn_global_load_async_to_lds_b128) && \
    __has_builtin(__builtin_amdgcn_s_wait_asynccnt)
  for (int i = tid; i < nvec; i += (int)blockDim.x) {
    int r = i / vpr, c = i - r * vpr;
    const bf16_t* g = Ab + (long long)r * lda + c * 8;
    // AS1/AS3 int4 pointers built via integer casts (low 32 bits of a generic
    // LDS address are the LDS byte offset per the aperture encoding).
    __builtin_amdgcn_global_load_async_to_lds_b128(
        (gv4i_t*)(uintptr_t)g, (lv4i_t*)(unsigned)(uintptr_t)(As + i * 8), 0,
        0);
  }
  __builtin_amdgcn_s_wait_asynccnt(0);
#else
  for (int i = tid; i < nvec; i += (int)blockDim.x) {
    int r = i / vpr, c = i - r * vpr;
    *(v8bf*)(As + i * 8) = *(const v8bf*)(Ab + (long long)r * lda + c * 8);
  }
#endif
  __syncthreads();

  int group = bg * WPB + wave;  // 4 consecutive n-tiles
  if (group * 64 >= N) return;

  const bf16_t* arow0 = As + (lane & 15) * K;
  const bf16_t* arow1 = As + (16 + (lane & 15)) * K;
  const bf16_t* wrow0 =
      W + batch * sW + (long long)(group * 64 + (lane & 15)) * ldw;

  v8f z = {0.f, 0.f, 0.f, 0.f, 0.f, 0.f, 0.f, 0.f};
  v8f acc[8];
#pragma unroll
  for (int i = 0; i < 8; ++i) acc[i] = z;

  for (int k0 = 0; k0 < K; k0 += 32) {
    v16bf a0 = load_fragA(arow0, k0, lane);  // ds_load_b128 x2
    v16bf a1 = load_fragA(arow1, k0, lane);
#pragma unroll
    for (int j = 0; j < 4; ++j) {
      v16bf bv = load_fragB(wrow0 + (long long)j * 16 * ldw, k0, lane);
      acc[j] = __builtin_amdgcn_wmma_f32_16x16x32_bf16(false, a0, false, bv,
                                                       (short)0, acc[j], false,
                                                       false);
      acc[4 + j] = __builtin_amdgcn_wmma_f32_16x16x32_bf16(
          false, a1, false, bv, (short)0, acc[4 + j], false, false);
    }
  }

  float* outp = Out + batch * sO;
  bf16_t* outbf = OutBf ? (OutBf + batch * sOB) : (bf16_t*)0;
#pragma unroll
  for (int mi = 0; mi < 2; ++mi) {
    int mbase = mi * 16 + ((lane & 16) ? 8 : 0);
#pragma unroll
    for (int j = 0; j < 4; ++j) {
      int n = group * 64 + j * 16 + (lane & 15);
      float bb = bias ? bias[batch * sBias + n] : 0.f;
#pragma unroll
      for (int r = 0; r < 8; ++r) {
        float v = actf(acc[mi * 4 + j][r] + bb, act);
        long long m = mbase + r;
        outp[m * ldo + n] = v;
        if (outbf) outbf[m * ldob + n] = (bf16_t)v;
      }
    }
  }
}

// ---------------------------------------------------------------------------
// Elementwise / gather helpers
// ---------------------------------------------------------------------------
__global__ void cvt_bf16_kernel(const float* __restrict__ in,
                                bf16_t* __restrict__ out, long long n) {
  long long i = (long long)blockIdx.x * blockDim.x + threadIdx.x;
  long long st = (long long)gridDim.x * blockDim.x;
  for (; i < n; i += st) out[i] = (bf16_t)in[i];
}

__global__ void copy_f32_kernel(const float* __restrict__ in,
                                float* __restrict__ out, long long n) {
  long long i = (long long)blockIdx.x * blockDim.x + threadIdx.x;
  long long st = (long long)gridDim.x * blockDim.x;
  for (; i < n; i += st) out[i] = in[i];
}

__global__ void cvt_strided_kernel(const float* __restrict__ src, int lds,
                                   int soff, bf16_t* __restrict__ dst, int ldd,
                                   int doff, int rows, int cols) {
  int i = blockIdx.x * blockDim.x + threadIdx.x;
  if (i >= rows * cols) return;
  int r = i / cols, c = i - r * cols;
  dst[(long long)r * ldd + doff + c] = (bf16_t)src[(long long)r * lds + soff + c];
}

// emb[t*B+b, e] = C[in_tok(t,b), e] ; in_tok = SOS for t==0 else target[b,t-1]
__global__ void embed_kernel(const float* __restrict__ Cw,
                             const int* __restrict__ tgt,
                             bf16_t* __restrict__ emb) {
  int i = blockIdx.x * blockDim.x + threadIdx.x;
  if (i >= T * B * E) return;
  int e = i % E;
  int tb = i / E;
  int b = tb % B;
  int t = tb / B;
  int tok = (t == 0) ? SOS : tgt[b * T + (t - 1)];
  emb[i] = (bf16_t)Cw[(long long)tok * E + e];
}

// m_ext[b,s,h] = ext_mem[b,s,h] * global_pointer[b,s]
__global__ void mext_kernel(const float* __restrict__ ext,
                            const float* __restrict__ gp,
                            float* __restrict__ out) {
  long long n = (long long)B * S * H;
  long long i = (long long)blockIdx.x * blockDim.x + threadIdx.x;
  long long st = (long long)gridDim.x * blockDim.x;
  for (; i < n; i += st) out[i] = ext[i] * gp[i / H];
}

// Broadcast h0 into the 4 GRU state slots (fp32 + bf16)
__global__ void replicate_kernel(const float* __restrict__ h0,
                                 float* __restrict__ hf,
                                 bf16_t* __restrict__ hb) {
  int i = blockIdx.x * blockDim.x + threadIdx.x;
  if (i >= 4 * B * H) return;
  float v = h0[i % (B * H)];
  hf[i] = v;
  hb[i] = (bf16_t)v;
}

// GRU gate math.  gi_all: [4, T*B, 3H] (includes bi); gh: [4, B, 3H] (incl bh)
__global__ void gru_update_kernel(const float* __restrict__ gi_all,
                                  const float* __restrict__ gh,
                                  float* __restrict__ hf,
                                  bf16_t* __restrict__ hb, int t) {
  int i = blockIdx.x * blockDim.x + threadIdx.x;
  if (i >= 4 * B * H) return;
  int j = i % H;
  int b = (i / H) % B;
  int d = i / (H * B);
  const float* gi =
      gi_all + ((long long)d * (T * B) + (long long)t * B + b) * H3;
  const float* g = gh + (long long)(d * B + b) * H3;
  float r = 1.f / (1.f + __expf(-(gi[j] + g[j])));
  float z = 1.f / (1.f + __expf(-(gi[j + H] + g[j + H])));
  float nn = tanhf(gi[j + 2 * H] + r * g[j + 2 * H]);
  float h = hf[i];
  float hn = (1.f - z) * nn + z * h;
  hf[i] = hn;
  hb[i] = (bf16_t)hn;
}

// Mix softmax over 3 local domains + assemble qin = [hg, h_local] in bf16.
__global__ void mix_kernel(const float* __restrict__ hf,
                           const float* __restrict__ mix_w,
                           const float* __restrict__ mix_b,
                           bf16_t* __restrict__ qin) {
  int b = blockIdx.x, tid = threadIdx.x;
  __shared__ float r0[H], r1[H], r2[H];
  __shared__ float sc[3];
  float h1 = hf[(1 * B + b) * H + tid];
  float h2 = hf[(2 * B + b) * H + tid];
  float h3 = hf[(3 * B + b) * H + tid];
  r0[tid] = h1 * mix_w[0 * H3 + tid * 3 + 0] + h2 * mix_w[0 * H3 + tid * 3 + 1] +
            h3 * mix_w[0 * H3 + tid * 3 + 2];
  r1[tid] = h1 * mix_w[1 * H3 + tid * 3 + 0] + h2 * mix_w[1 * H3 + tid * 3 + 1] +
            h3 * mix_w[1 * H3 + tid * 3 + 2];
  r2[tid] = h1 * mix_w[2 * H3 + tid * 3 + 0] + h2 * mix_w[2 * H3 + tid * 3 + 1] +
            h3 * mix_w[2 * H3 + tid * 3 + 2];
  __syncthreads();
  for (int off = H / 2; off > 0; off >>= 1) {
    if (tid < off) {
      r0[tid] += r0[tid + off];
      r1[tid] += r1[tid + off];
      r2[tid] += r2[tid + off];
    }
    __syncthreads();
  }
  if (tid == 0) {
    float s0 = r0[0] + mix_b[0], s1 = r1[0] + mix_b[1], s2 = r2[0] + mix_b[2];
    float m = fmaxf(s0, fmaxf(s1, s2));
    float e0 = __expf(s0 - m), e1 = __expf(s1 - m), e2 = __expf(s2 - m);
    float inv = 1.f / (e0 + e1 + e2);
    sc[0] = e0 * inv; sc[1] = e1 * inv; sc[2] = e2 * inv;
  }
  __syncthreads();
  float hl = h1 * sc[0] + h2 * sc[1] + h3 * sc[2];
  qin[b * (2 * H) + tid] = (bf16_t)hf[(0 * B + b) * H + tid];  // hg
  qin[b * (2 * H) + H + tid] = (bf16_t)hl;                     // h_local
}

// Additive attention: s[l] = sum_g tanh(prem[b,l,g]+qproj[b,g])*w2[g] + b2;
// softmax over l; out[b,:] = sum_l a[l]*mem[b,l,:]  (bf16 into concat slice)
__global__ void addattn_kernel(const float* __restrict__ prem,
                               const float* __restrict__ qproj,
                               const float* __restrict__ w2,
                               const float* __restrict__ b2,
                               const float* __restrict__ mem, int L,
                               bf16_t* __restrict__ outbf, int ldob) {
  int b = blockIdx.x, tid = threadIdx.x;  // H threads
  __shared__ float sqp[H], sw2[H], red[H], prob[H];
  sqp[tid] = qproj[b * H + tid];
  sw2[tid] = w2[tid];
  __syncthreads();
  float s = -1e30f;
  if (tid < L) {
    const float* pr = prem + ((long long)b * L + tid) * H;
    float acc = 0.f;
    for (int g = 0; g < H; ++g) acc += tanhf(pr[g] + sqp[g]) * sw2[g];
    s = acc + b2[0];
  }
  red[tid] = s;
  __syncthreads();
  for (int off = H / 2; off > 0; off >>= 1) {
    if (tid < off) red[tid] = fmaxf(red[tid], red[tid + off]);
    __syncthreads();
  }
  float mx = red[0];
  __syncthreads();
  float ex = (tid < L) ? __expf(s - mx) : 0.f;
  red[tid] = ex;
  __syncthreads();
  for (int off = H / 2; off > 0; off >>= 1) {
    if (tid < off) red[tid] += red[tid + off];
    __syncthreads();
  }
  float inv = 1.f / red[0];
  prob[tid] = ex * inv;
  __syncthreads();
  float o = 0.f;
  for (int l = 0; l < L; ++l)
    o += prob[l] * mem[((long long)b * L + l) * H + tid];
  outbf[b * ldob + tid] = (bf16_t)o;
}

// ptr[b,s] = sum_h m_ext[b,s,h] * ctx2[b,h]
__global__ void ptr_kernel(const float* __restrict__ mext,
                           const float* __restrict__ ctx2,
                           float* __restrict__ out) {
  int b = blockIdx.x, s = threadIdx.x;  // S threads
  __shared__ float sc[H];
  sc[s] = ctx2[b * H + s];
  __syncthreads();
  const float* m = mext + ((long long)b * S + s) * H;
  float acc = 0.f;
  for (int h = 0; h < H; ++h) acc += m[h] * sc[h];
  out[b * S + s] = acc;
}

// ---------------------------------------------------------------------------
static void launch_gemm(hipStream_t st, const bf16_t* A, long long sA, int lda,
                        const bf16_t* W, long long sW, int ldw,
                        const float* bias, long long sBias, float* Out,
                        long long sO, int ldo, bf16_t* OutBf, long long sOB,
                        int ldob, int M, int N, int K, int act, int nb) {
  int waves = nb * (M / 32) * (N / 64);
  int blocks = (waves + 7) / 8;  // 8 waves (256 thr) per block
  wmma_gemm_kernel<<<blocks, 256, 0, st>>>(A, sA, lda, W, sW, ldw, bias, sBias,
                                           Out, sO, ldo, OutBf, sOB, ldob, M, N,
                                           K, act, waves);
}

static void launch_gemm_m32(hipStream_t st, const bf16_t* A, long long sA,
                            int lda, const bf16_t* W, long long sW, int ldw,
                            const float* bias, long long sBias, float* Out,
                            long long sO, int ldo, bf16_t* OutBf, long long sOB,
                            int ldob, int N, int K, int act, int nb) {
  int groups = N / 64;
  int bpb = (groups + WPB - 1) / WPB;
  size_t lds = (size_t)32 * K * sizeof(bf16_t);
  wmma_gemm_m32_kernel<<<nb * bpb, WPB * 32, lds, st>>>(
      A, sA, lda, W, sW, ldw, bias, sBias, Out, sO, ldo, OutBf, sOB, ldob, N, K,
      act, bpb);
}

extern "C" void kernel_launch(void* const* d_in, const int* in_sizes, int n_in,
                              void* d_out, int out_size, void* d_ws,
                              size_t ws_size, hipStream_t stream) {
  (void)in_sizes; (void)n_in; (void)out_size; (void)ws_size;
  const float* encode_hidden  = (const float*)d_in[0];
  const float* H_mem          = (const float*)d_in[1];
  const float* kb_output      = (const float*)d_in[2];
  const float* kb_emb         = (const float*)d_in[3];
  const float* kb_readout     = (const float*)d_in[4];
  const float* ext_mem        = (const float*)d_in[5];
  const float* global_pointer = (const float*)d_in[6];
  const int*   targets        = (const int*)d_in[7];
  const float* Cw    = (const float*)d_in[8];
  const float* proj_w = (const float*)d_in[9];
  const float* proj_b = (const float*)d_in[10];
  const float* gg_Wi = (const float*)d_in[11];
  const float* gg_Wh = (const float*)d_in[12];
  const float* gg_bi = (const float*)d_in[13];
  const float* gg_bh = (const float*)d_in[14];
  const float* gl_Wi = (const float*)d_in[15];
  const float* gl_Wh = (const float*)d_in[16];
  const float* gl_bi = (const float*)d_in[17];
  const float* gl_bh = (const float*)d_in[18];
  const float* mix_w = (const float*)d_in[19];
  const float* mix_b = (const float*)d_in[20];
  const float* mlp_w = (const float*)d_in[21];
  const float* mlp_b = (const float*)d_in[22];
  const float* at_w1 = (const float*)d_in[23];
  const float* at_w2 = (const float*)d_in[24];
  const float* at_b2 = (const float*)d_in[25];
  const float* atf_w1 = (const float*)d_in[26];
  const float* atf_w2 = (const float*)d_in[27];
  const float* atf_b2 = (const float*)d_in[28];
  const float* ake_w1 = (const float*)d_in[29];
  const float* ake_w2 = (const float*)d_in[30];
  const float* ake_b2 = (const float*)d_in[31];
  const float* p3_w = (const float*)d_in[32];
  const float* p3_b = (const float*)d_in[33];
  const float* p4_w = (const float*)d_in[34];
  const float* p4_b = (const float*)d_in[35];
  float* out = (float*)d_out;

  // ---- workspace carving (all offsets 256B aligned) ----
  char* ws = (char*)d_ws;
  size_t off = 0;
  auto alloc = [&](size_t bytes) -> void* {
    void* p = ws + off;
    off = (off + bytes + 255) & ~(size_t)255;
    return p;
  };
  bf16_t* C_bf     = (bf16_t*)alloc((size_t)V * E * 2);
  bf16_t* Wi_bf    = (bf16_t*)alloc((size_t)4 * H3 * H * 2);
  bf16_t* Wh_bf    = (bf16_t*)alloc((size_t)4 * H3 * H * 2);
  float*  bi_cat   = (float*)alloc((size_t)4 * H3 * 4);
  float*  bh_cat   = (float*)alloc((size_t)4 * H3 * 4);
  bf16_t* emb_bf   = (bf16_t*)alloc((size_t)T * B * E * 2);
  float*  gi_all   = (float*)alloc((size_t)4 * T * B * H3 * 4);
  bf16_t* mlp_bf   = (bf16_t*)alloc((size_t)H * 2 * H * 2);
  bf16_t* p3_bf    = (bf16_t*)alloc((size_t)H * H3 * 2);
  bf16_t* p4_bf    = (bf16_t*)alloc((size_t)H * H3 * 2);
  bf16_t* proj_bf  = (bf16_t*)alloc((size_t)H * 2 * H * 2);
  bf16_t* enc_bf   = (bf16_t*)alloc((size_t)B * 2 * H * 2);
  bf16_t* atm_bf   = (bf16_t*)alloc((size_t)H * H * 2);
  bf16_t* atfm_bf  = (bf16_t*)alloc((size_t)H * H * 2);
  bf16_t* akem_bf  = (bf16_t*)alloc((size_t)H * H * 2);
  bf16_t* atq2_bf  = (bf16_t*)alloc((size_t)2 * H * H * 2);  // [at_q ; atf_q]
  bf16_t* akeq_bf  = (bf16_t*)alloc((size_t)H * H * 2);
  bf16_t* Hmem_bf  = (bf16_t*)alloc((size_t)B * LH * H * 2);
  bf16_t* kbo_bf   = (bf16_t*)alloc((size_t)B * LKB * H * 2);
  bf16_t* kbe_bf   = (bf16_t*)alloc((size_t)B * LKE * H * 2);
  float*  prem_at  = (float*)alloc((size_t)B * LH * H * 4);
  float*  prem_kb  = (float*)alloc((size_t)B * LKB * H * 4);
  float*  prem_ake = (float*)alloc((size_t)B * LKE * H * 4);
  float*  m_ext    = (float*)alloc((size_t)B * S * H * 4);
  float*  hf       = (float*)alloc((size_t)4 * B * H * 4);
  bf16_t* hb       = (bf16_t*)alloc((size_t)4 * B * H * 2);
  float*  gh_buf   = (float*)alloc((size_t)4 * B * H3 * 4);
  bf16_t* qin_bf   = (bf16_t*)alloc((size_t)B * 2 * H * 2);
  bf16_t* ctxin_bf = (bf16_t*)alloc((size_t)B * H3 * 2);   // [sket|q|kbh]
  bf16_t* ctx2in_bf= (bf16_t*)alloc((size_t)B * H3 * 2);   // [out_h|ctx|kb_ro]
  float*  q_scr    = (float*)alloc((size_t)B * H * 4);
  float*  ctx_scr  = (float*)alloc((size_t)B * H * 4);
  float*  ctx2_f   = (float*)alloc((size_t)B * H * 4);
  float*  qproj2   = (float*)alloc((size_t)2 * B * H * 4);
  float*  qproj_ake= (float*)alloc((size_t)B * H * 4);
  float*  h0       = (float*)alloc((size_t)B * H * 4);

  auto cvt = [&](const float* src, bf16_t* dst, long long n) {
    int blocks = (int)((n + 255) / 256);
    if (blocks > 4096) blocks = 4096;
    cvt_bf16_kernel<<<blocks, 256, 0, stream>>>(src, dst, n);
  };
  auto copyf = [&](const float* src, float* dst, long long n) {
    int blocks = (int)((n + 255) / 256);
    copy_f32_kernel<<<blocks, 256, 0, stream>>>(src, dst, n);
  };
  auto cvts = [&](const float* src, int lds, int soff, bf16_t* dst, int ldd,
                  int doff, int rows, int cols) {
    int n = rows * cols;
    cvt_strided_kernel<<<(n + 255) / 256, 256, 0, stream>>>(src, lds, soff, dst,
                                                            ldd, doff, rows, cols);
  };

  // ---- one-time setup ----
  cvt(Cw, C_bf, (long long)V * E);
  cvt(gg_Wi, Wi_bf, (long long)H3 * H);
  cvt(gl_Wi, Wi_bf + (long long)H3 * H, (long long)3 * H3 * H);
  cvt(gg_Wh, Wh_bf, (long long)H3 * H);
  cvt(gl_Wh, Wh_bf + (long long)H3 * H, (long long)3 * H3 * H);
  cvt(mlp_w, mlp_bf, (long long)H * 2 * H);
  cvt(p3_w, p3_bf, (long long)H * H3);
  cvt(p4_w, p4_bf, (long long)H * H3);
  cvt(proj_w, proj_bf, (long long)H * 2 * H);
  cvt(encode_hidden, enc_bf, (long long)B * 2 * H);
  cvt(H_mem, Hmem_bf, (long long)B * LH * H);
  cvt(kb_output, kbo_bf, (long long)B * LKB * H);
  cvt(kb_emb, kbe_bf, (long long)B * LKE * H);
  copyf(gg_bi, bi_cat, H3);
  copyf(gl_bi, bi_cat + H3, 3 * H3);
  copyf(gg_bh, bh_cat, H3);
  copyf(gl_bh, bh_cat + H3, 3 * H3);
  cvts(at_w1, 2 * H, 0, atm_bf, H, 0, H, H);
  cvts(at_w1, 2 * H, H, atq2_bf, H, 0, H, H);
  cvts(atf_w1, 2 * H, 0, atfm_bf, H, 0, H, H);
  cvts(atf_w1, 2 * H, H, atq2_bf + H * H, H, 0, H, H);
  cvts(ake_w1, 2 * H, 0, akem_bf, H, 0, H, H);
  cvts(ake_w1, 2 * H, H, akeq_bf, H, 0, H, H);
  cvts(kb_readout, H, 0, ctx2in_bf, H3, 2 * H, B, H);  // constant slice

  embed_kernel<<<(T * B * E + 255) / 256, 256, 0, stream>>>(Cw, targets, emb_bf);
  mext_kernel<<<4096, 256, 0, stream>>>(ext_mem, global_pointer, m_ext);

  // gi_all[d] = emb @ Wi[d]^T + bi[d]   (A shared across 4 batches)
  launch_gemm(stream, emb_bf, 0, E, Wi_bf, (long long)H3 * H, H, bi_cat, H3,
              gi_all, (long long)T * B * H3, H3, nullptr, 0, 0, T * B, H3, E, 0, 4);
  // memory-side attention projections (no bias, no activation)
  launch_gemm(stream, Hmem_bf, 0, H, atm_bf, 0, H, nullptr, 0, prem_at, 0, H,
              nullptr, 0, 0, B * LH, H, H, 0, 1);
  launch_gemm(stream, kbo_bf, 0, H, atfm_bf, 0, H, nullptr, 0, prem_kb, 0, H,
              nullptr, 0, 0, B * LKB, H, H, 0, 1);
  launch_gemm(stream, kbe_bf, 0, H, akem_bf, 0, H, nullptr, 0, prem_ake, 0, H,
              nullptr, 0, 0, B * LKE, H, H, 0, 1);
  // h0 = relu(encode_hidden @ proj_w^T + proj_b)
  launch_gemm_m32(stream, enc_bf, 0, 2 * H, proj_bf, 0, 2 * H, proj_b, 0, h0,
                  0, H, nullptr, 0, 0, H, 2 * H, 2, 1);
  replicate_kernel<<<(4 * B * H + 255) / 256, 256, 0, stream>>>(h0, hf, hb);

  // ---- sequential decode ----
  for (int t = 0; t < T; ++t) {
    // gh[d] = h[d] @ Wh[d]^T + bh[d]
    launch_gemm_m32(stream, hb, (long long)B * H, H, Wh_bf, (long long)H3 * H,
                    H, bh_cat, H3, gh_buf, (long long)B * H3, H3, nullptr, 0, 0,
                    H3, H, 0, 4);
    gru_update_kernel<<<(4 * B * H + 255) / 256, 256, 0, stream>>>(gi_all,
                                                                   gh_buf, hf,
                                                                   hb, t);
    mix_kernel<<<B, H, 0, stream>>>(hf, mix_w, mix_b, qin_bf);
    // q = qin @ mlp_w^T + mlp_b  (bf16 copy -> ctxin cols [H,2H))
    launch_gemm_m32(stream, qin_bf, 0, 2 * H, mlp_bf, 0, 2 * H, mlp_b, 0,
                    q_scr, 0, H, ctxin_bf + H, 0, H3, H, 2 * H, 0, 1);
    // query projections for sketch + kb attentions (batched, shared A = q)
    launch_gemm_m32(stream, ctxin_bf + H, 0, H3, atq2_bf, (long long)H * H, H,
                    nullptr, 0, qproj2, (long long)B * H, H, nullptr, 0, 0, H,
                    H, 0, 2);
    addattn_kernel<<<B, H, 0, stream>>>(prem_at, qproj2, at_w2, at_b2, H_mem,
                                        LH, ctxin_bf, H3);            // sket
    addattn_kernel<<<B, H, 0, stream>>>(prem_kb, qproj2 + B * H, atf_w2, atf_b2,
                                        kb_output, LKB, ctxin_bf + 2 * H, H3);
    // ctx = tanh(ctxin @ p4^T + p4_b)  (bf16 copy -> ctx2in cols [H,2H))
    launch_gemm_m32(stream, ctxin_bf, 0, H3, p4_bf, 0, H3, p4_b, 0, ctx_scr, 0,
                    H, ctx2in_bf + H, 0, H3, H, H3, 1, 1);
    // ake query projection (A = ctx slice of ctx2in)
    launch_gemm_m32(stream, ctx2in_bf + H, 0, H3, akeq_bf, 0, H, nullptr, 0,
                    qproj_ake, 0, H, nullptr, 0, 0, H, H, 0, 1);
    addattn_kernel<<<B, H, 0, stream>>>(prem_ake, qproj_ake, ake_w2, ake_b2,
                                        kb_emb, LKE, ctx2in_bf, H3);  // out_h
    // ctx2 = tanh(ctx2in @ p3^T + p3_b)
    launch_gemm_m32(stream, ctx2in_bf, 0, H3, p3_bf, 0, H3, p3_b, 0, ctx2_f, 0,
                    H, nullptr, 0, 0, H, H3, 1, 1);
    // vocab logits: ctx @ C^T  -> out[t]
    launch_gemm_m32(stream, ctx2in_bf + H, 0, H3, C_bf, 0, E, nullptr, 0,
                    out + (long long)t * B * V, 0, V, nullptr, 0, 0, V, E, 0, 1);
    // pointer logits
    ptr_kernel<<<B, S, 0, stream>>>(
        m_ext, ctx2_f, out + (long long)T * B * V + (long long)t * B * S);
  }
}
